// JambaBlock_35734127903109
// MI455X (gfx1250) — matrix operand order repaired and checked by reading, not compiled, not run on hardware
//
#include <hip/hip_runtime.h>
#include <hip/hip_bf16.h>

// ---------------------------------------------------------------------------
// Jamba block for gfx1250 (MI455X). All GEMMs via v_wmma_f32_16x16x32_bf16.
// ---------------------------------------------------------------------------

typedef __attribute__((ext_vector_type(16))) __bf16 bf16x16;
typedef __attribute__((ext_vector_type(8)))  __bf16 bf16x8;
typedef __attribute__((ext_vector_type(4)))  __bf16 bf16x4;
typedef __attribute__((ext_vector_type(8)))  float  f32x8;
typedef __attribute__((ext_vector_type(4)))  float  f32x4;

#define S_LEN   1024
#define DIM_    1024
#define D_INNER 2048
#define D_STATE 16
#define DT_RANK 64
#define HID_    4096
#define NEXP    8
#define NHEADS  16
#define HDIM    64

__device__ __forceinline__ float siluf(float v) { return v / (1.f + __expf(-v)); }
__device__ __forceinline__ float geluf(float v) {
    const float c = 0.7978845608028654f;
    float t = tanhf(c * (v + 0.044715f * v * v * v));
    return 0.5f * v * (1.f + t);
}
__device__ __forceinline__ float softplusf(float v) {
    return (v > 20.f) ? v : log1pf(__expf(v));
}
__device__ __forceinline__ float applyact(float v, int act) {
    if (act == 1) return siluf(v);
    if (act == 2) return geluf(v);
    if (act == 3) return softplusf(v);
    return v;
}

// ---------------------------------------------------------------------------
// WMMA bf16 GEMM: C[M,N] = act(A[M,K] @ B[K,N] + bias). Assumptions (hold for
// every call site here): M % 64 == 0, K % 64 == 0. Only N may be ragged.
//
// Block = 256 threads (8 wave32) computing a 64x128 tile; waves form a 2x4
// grid, each wave owns a 32x32 quadrant = 4 accumulators -> 4 A + 4 B
// ds_load_b128 per 4 WMMAs (2:1), 8 v_wmma per barrier pair (BK=64).
// LDS: A row-major (m, k-contig), B transposed (n, k-contig); rows padded to
// 72 bf16 (144B, 16B aligned) so every fragment read is a ds_load_b128.
// ---------------------------------------------------------------------------
#define BM 64
#define BN 128
#define BK 64
#define LDSS 72

__global__ __launch_bounds__(256) void gemm_wmma_bf16(
    const float* __restrict__ A, int lda,
    const float* __restrict__ Bw, int ldb,
    float* __restrict__ C, int ldc,
    const float* __restrict__ bias,
    int M, int N, int K, int act)
{
    __shared__ __bf16 As[BM * LDSS];
    __shared__ __bf16 Bs[BN * LDSS];
    const int tid  = threadIdx.x;
    const int wave = tid >> 5;
    const int lane = tid & 31;
    const int l15  = lane & 15;
    const int hi   = lane >> 4;              // 0 or 1
    const int bm   = blockIdx.y * BM;
    const int bn   = blockIdx.x * BN;
    const int wm   = (wave & 1) * 32;        // wave's M quadrant (32 rows)
    const int wn   = (wave >> 1) * 32;       // wave's N quadrant (32 cols)

    const bool fullN = (bn + BN) <= N;       // uniform per block

    f32x8 acc00 = {}, acc01 = {}, acc10 = {}, acc11 = {};

    for (int k0 = 0; k0 < K; k0 += BK) {
        if (fullN) {
            // ---- fast path: unguarded vector loads, batched ----
            f32x4 a_reg[4], b_reg[8];
            #pragma unroll
            for (int i = 0; i < 4; ++i) {        // A: 64x64 = 1024 float4
                int idx = i * 256 + tid;
                int r = idx >> 4, c4 = (idx & 15) << 2;
                a_reg[i] = *(const f32x4*)(A + (long)(bm + r) * lda + (k0 + c4));
            }
            #pragma unroll
            for (int i = 0; i < 8; ++i) {        // B: 64x128 = 2048 float4
                int idx = i * 256 + tid;
                int kk = idx >> 5, n4 = (idx & 31) << 2;
                b_reg[i] = *(const f32x4*)(Bw + (long)(k0 + kk) * ldb + (bn + n4));
            }
            #pragma unroll
            for (int i = 0; i < 4; ++i) {
                int idx = i * 256 + tid;
                int r = idx >> 4, c4 = (idx & 15) << 2;
                bf16x4 cv;
                cv[0] = (__bf16)a_reg[i][0]; cv[1] = (__bf16)a_reg[i][1];
                cv[2] = (__bf16)a_reg[i][2]; cv[3] = (__bf16)a_reg[i][3];
                *(bf16x4*)(&As[r * LDSS + c4]) = cv;   // ds_store_b64
            }
            #pragma unroll
            for (int i = 0; i < 8; ++i) {
                int idx = i * 256 + tid;
                int kk = idx >> 5, n4 = (idx & 31) << 2;
                #pragma unroll
                for (int j = 0; j < 4; ++j)
                    Bs[(n4 + j) * LDSS + kk] = (__bf16)b_reg[i][j];
            }
        } else {
            // ---- edge blocks (ragged N: 96/64/8) only: guarded staging ----
            #pragma unroll
            for (int i = 0; i < 16; ++i) {       // A: 64x64 elems
                int idx = i * 256 + tid;
                int r = idx >> 6, c = idx & 63;
                As[r * LDSS + c] = (__bf16)A[(long)(bm + r) * lda + (k0 + c)];
            }
            #pragma unroll
            for (int i = 0; i < 32; ++i) {       // B: 64x128 elems
                int idx = i * 256 + tid;
                int kk = idx >> 7, nn = idx & 127;
                int gn = bn + nn;
                float v = (gn < N) ? Bw[(long)(k0 + kk) * ldb + gn] : 0.f;
                Bs[nn * LDSS + kk] = (__bf16)v;
            }
        }
        __syncthreads();

        #pragma unroll
        for (int ks = 0; ks < BK; ks += 32) {
            // A fragments (rows wm+l15 and wm+16+l15):
            // K runs {ks+hi*8..+7} and {ks+16+hi*8..+7}
            bf16x16 af0, af1;
            {
                bf16x8 a0 = *(const bf16x8*)(&As[(wm + l15) * LDSS + ks + hi * 8]);
                bf16x8 a1 = *(const bf16x8*)(&As[(wm + l15) * LDSS + ks + 16 + hi * 8]);
                #pragma unroll
                for (int i = 0; i < 8; ++i) { af0[i] = a0[i]; af0[8 + i] = a1[i]; }
                bf16x8 a2 = *(const bf16x8*)(&As[(wm + 16 + l15) * LDSS + ks + hi * 8]);
                bf16x8 a3 = *(const bf16x8*)(&As[(wm + 16 + l15) * LDSS + ks + 16 + hi * 8]);
                #pragma unroll
                for (int i = 0; i < 8; ++i) { af1[i] = a2[i]; af1[8 + i] = a3[i]; }
            }
            // B fragments (cols wn+l15 and wn+16+l15):
            // K runs {ks+hi*8..+7} and {ks+16+hi*8..+7}
            bf16x16 bf0, bf1;
            {
                const __bf16* bp0 = &Bs[(wn + l15) * LDSS + ks + hi * 8];
                bf16x8 b0a = *(const bf16x8*)bp0;
                bf16x8 b0b = *(const bf16x8*)(bp0 + 16);
                #pragma unroll
                for (int i = 0; i < 8; ++i) { bf0[i] = b0a[i]; bf0[8 + i] = b0b[i]; }
                const __bf16* bp1 = &Bs[(wn + 16 + l15) * LDSS + ks + hi * 8];
                bf16x8 b1a = *(const bf16x8*)bp1;
                bf16x8 b1b = *(const bf16x8*)(bp1 + 16);
                #pragma unroll
                for (int i = 0; i < 8; ++i) { bf1[i] = b1a[i]; bf1[8 + i] = b1b[i]; }
            }

            acc00 = __builtin_amdgcn_wmma_f32_16x16x32_bf16(
                        false, af0, false, bf0, (short)0, acc00, false, false);
            acc01 = __builtin_amdgcn_wmma_f32_16x16x32_bf16(
                        false, af0, false, bf1, (short)0, acc01, false, false);
            acc10 = __builtin_amdgcn_wmma_f32_16x16x32_bf16(
                        false, af1, false, bf0, (short)0, acc10, false, false);
            acc11 = __builtin_amdgcn_wmma_f32_16x16x32_bf16(
                        false, af1, false, bf1, (short)0, acc11, false, false);
        }
        __syncthreads();
    }

    // epilogue: C/D layout -> element r = row (r + hi*8), col l15. M always
    // tile-aligned here; guard N only.
    #pragma unroll
    for (int r = 0; r < 8; ++r) {
        int gm0 = bm + wm + r + hi * 8;
        int gm1 = gm0 + 16;
        int gn0 = bn + wn + l15;
        int gn1 = gn0 + 16;
        if (gn0 < N) {
            float b = bias ? bias[gn0] : 0.f;
            C[(long)gm0 * ldc + gn0] = applyact(acc00[r] + b, act);
            C[(long)gm1 * ldc + gn0] = applyact(acc10[r] + b, act);
        }
        if (gn1 < N) {
            float b = bias ? bias[gn1] : 0.f;
            C[(long)gm0 * ldc + gn1] = applyact(acc01[r] + b, act);
            C[(long)gm1 * ldc + gn1] = applyact(acc11[r] + b, act);
        }
    }
}

// ---------------------------------------------------------------------------
// Glue kernels
// ---------------------------------------------------------------------------
__global__ void rms_kernel(const float* __restrict__ x, float* __restrict__ y)
{
    int row = blockIdx.x;
    __shared__ float red[256];
    float s = 0.f;
    for (int i = threadIdx.x; i < DIM_; i += 256) {
        float v = x[row * DIM_ + i]; s += v * v;
    }
    red[threadIdx.x] = s; __syncthreads();
    for (int st = 128; st > 0; st >>= 1) {
        if (threadIdx.x < st) red[threadIdx.x] += red[threadIdx.x + st];
        __syncthreads();
    }
    float n = sqrtf(red[0]);
    float sc = 32.f / fmaxf(n, 1e-12f);      // sqrt(DIM)=32
    for (int i = threadIdx.x; i < DIM_; i += 256)
        y[row * DIM_ + i] = x[row * DIM_ + i] * sc;
}

__global__ void add_kernel(float* __restrict__ dst, const float* __restrict__ a,
                           const float* __restrict__ b, int n)
{
    int i = blockIdx.x * 256 + threadIdx.x;
    if (i < n) dst[i] = a[i] + b[i];
}

// depthwise causal conv (D_CONV=4) + bias + silu; reads xi half of xz
__global__ void conv_silu_kernel(const float* __restrict__ xz,
                                 const float* __restrict__ cw,
                                 const float* __restrict__ cb,
                                 float* __restrict__ xi)
{
    int idx = blockIdx.x * 256 + threadIdx.x;      // S*D_INNER
    int t = idx >> 11, d = idx & (D_INNER - 1);
    float acc = cb[d];
    #pragma unroll
    for (int j = 0; j < 4; ++j) {
        int tt = t - 3 + j;
        if (tt >= 0) acc += xz[tt * (2 * D_INNER) + d] * cw[d * 4 + j];
    }
    xi[idx] = siluf(acc);
}

// selective scan: one thread per channel d, sequential over t, state in regs
__global__ __launch_bounds__(256) void scan_kernel(
    const float* __restrict__ u,      // [S, D_INNER]
    const float* __restrict__ delta,  // [S, D_INNER]
    const float* __restrict__ dbc,    // [S, 96]  (B at 64..79, C at 80..95)
    const float* __restrict__ A_log,  // [D_INNER, 16]
    const float* __restrict__ Dv,     // [D_INNER]
    float* __restrict__ y)            // [S, D_INNER]
{
    int d = blockIdx.x * 256 + threadIdx.x;
    __shared__ float sBC[32];
    float A[D_STATE];
    #pragma unroll
    for (int n = 0; n < D_STATE; ++n) A[n] = -__expf(A_log[d * D_STATE + n]);
    float h[D_STATE];
    #pragma unroll
    for (int n = 0; n < D_STATE; ++n) h[n] = 0.f;
    float Dd = Dv[d];
    for (int t = 0; t < S_LEN; ++t) {
        if (threadIdx.x < 32) sBC[threadIdx.x] = dbc[t * 96 + DT_RANK + threadIdx.x];
        __syncthreads();
        float dt = delta[t * D_INNER + d];
        float ut = u[t * D_INNER + d];
        float acc = 0.f;
        #pragma unroll
        for (int n = 0; n < D_STATE; ++n) {
            float dA = __expf(dt * A[n]);
            h[n] = dA * h[n] + dt * sBC[n] * ut;
            acc += h[n] * sBC[D_STATE + n];
        }
        y[t * D_INNER + d] = acc + ut * Dd;
        __syncthreads();
    }
}

// y *= silu(z), z = xz[:, D_INNER:]
__global__ void gatemul_kernel(float* __restrict__ y, const float* __restrict__ xz)
{
    int idx = blockIdx.x * 256 + threadIdx.x;
    int t = idx >> 11, d = idx & (D_INNER - 1);
    y[idx] *= siluf(xz[t * (2 * D_INNER) + D_INNER + d]);
}

// softmax over 8 experts + top-2 renormalized weights
__global__ void gate_topk_kernel(const float* __restrict__ gates, float* __restrict__ wts)
{
    int t = blockIdx.x * 256 + threadIdx.x;
    if (t >= S_LEN) return;
    float g[NEXP];
    float m = -1e30f;
    for (int e = 0; e < NEXP; ++e) { g[e] = gates[t * NEXP + e]; m = fmaxf(m, g[e]); }
    float sum = 0.f;
    for (int e = 0; e < NEXP; ++e) { g[e] = __expf(g[e] - m); sum += g[e]; }
    for (int e = 0; e < NEXP; ++e) g[e] /= sum;
    int i0 = 0;
    for (int e = 1; e < NEXP; ++e) if (g[e] > g[i0]) i0 = e;
    int i1 = (i0 == 0) ? 1 : 0;
    for (int e = 0; e < NEXP; ++e) if (e != i0 && g[e] > g[i1]) i1 = e;
    float tot = g[i0] + g[i1];
    for (int e = 0; e < NEXP; ++e) {
        float w = 0.f;
        if (e == i0) w = g[i0] / tot;
        else if (e == i1) w = g[i1] / tot;
        wts[t * NEXP + e] = w;
    }
}

__global__ void moe_accum_kernel(float* __restrict__ acc, const float* __restrict__ yv,
                                 const float* __restrict__ wts, int e, int first)
{
    int idx = blockIdx.x * 256 + threadIdx.x;      // S*DIM
    int s = idx >> 10;
    float v = wts[s * NEXP + e] * yv[idx];
    acc[idx] = first ? v : (acc[idx] + v);
}

// layernorm over HID per row, in place
__global__ void ln_kernel(float* __restrict__ h, const float* __restrict__ g,
                          const float* __restrict__ b)
{
    int row = blockIdx.x;
    __shared__ float red[256];
    float s = 0.f;
    for (int i = threadIdx.x; i < HID_; i += 256) s += h[row * HID_ + i];
    red[threadIdx.x] = s; __syncthreads();
    for (int st = 128; st > 0; st >>= 1) {
        if (threadIdx.x < st) red[threadIdx.x] += red[threadIdx.x + st];
        __syncthreads();
    }
    float mu = red[0] / HID_;
    __syncthreads();
    float v = 0.f;
    for (int i = threadIdx.x; i < HID_; i += 256) {
        float d = h[row * HID_ + i] - mu; v += d * d;
    }
    red[threadIdx.x] = v; __syncthreads();
    for (int st = 128; st > 0; st >>= 1) {
        if (threadIdx.x < st) red[threadIdx.x] += red[threadIdx.x + st];
        __syncthreads();
    }
    float inv = rsqrtf(red[0] / HID_ + 1e-5f);
    for (int i = threadIdx.x; i < HID_; i += 256)
        h[row * HID_ + i] = (h[row * HID_ + i] - mu) * inv * g[i] + b[i];
}

// MQA: one block per (query s, head h); K/V shared across heads (dim 64)
__global__ __launch_bounds__(256) void attn_kernel(
    const float* __restrict__ Q,   // [S, DIM] as (s, h*64+e)
    const float* __restrict__ Kc,  // [S, 64]
    const float* __restrict__ Vc,  // [S, 64]
    float* __restrict__ O)         // [S, DIM]
{
    int s = blockIdx.x, h = blockIdx.y;
    __shared__ float q[HDIM];
    __shared__ float p[S_LEN];
    __shared__ float red[256];
    int tid = threadIdx.x;
    if (tid < HDIM) q[tid] = Q[s * DIM_ + h * HDIM + tid];
    __syncthreads();
    const float scale = 0.125f;     // 1/sqrt(64)
    for (int t = tid; t < S_LEN; t += 256) {
        float acc = 0.f;
        #pragma unroll 8
        for (int e = 0; e < HDIM; ++e) acc += q[e] * Kc[t * HDIM + e];
        p[t] = acc * scale;
    }
    __syncthreads();
    float m = -1e30f;
    for (int t = tid; t < S_LEN; t += 256) m = fmaxf(m, p[t]);
    red[tid] = m; __syncthreads();
    for (int st = 128; st > 0; st >>= 1) {
        if (tid < st) red[tid] = fmaxf(red[tid], red[tid + st]);
        __syncthreads();
    }
    m = red[0]; __syncthreads();
    float sum = 0.f;
    for (int t = tid; t < S_LEN; t += 256) {
        float ev = __expf(p[t] - m); p[t] = ev; sum += ev;
    }
    red[tid] = sum; __syncthreads();
    for (int st = 128; st > 0; st >>= 1) {
        if (tid < st) red[tid] += red[tid + st];
        __syncthreads();
    }
    float inv = 1.f / red[0];
    __syncthreads();
    int grp = tid >> 6, e = tid & 63;
    float acc = 0.f;
    for (int t = grp * 256; t < (grp + 1) * 256; ++t) acc += p[t] * Vc[t * HDIM + e];
    red[grp * HDIM + e] = acc; __syncthreads();
    if (grp == 0)
        O[s * DIM_ + h * HDIM + e] = (red[e] + red[64 + e] + red[128 + e] + red[192 + e]) * inv;
}

// ---------------------------------------------------------------------------
// Host orchestration
// ---------------------------------------------------------------------------
struct MambaP {
    const float *in_w, *conv_w, *conv_b, *xproj_w, *dt_w, *dt_b, *A_log, *D, *out_w;
};

static void gemm(hipStream_t st, const float* A, int lda, const float* B, int ldb,
                 float* C, int ldc, const float* bias, int M, int N, int K, int act)
{
    dim3 grid((N + BN - 1) / BN, (M + BM - 1) / BM);
    gemm_wmma_bf16<<<grid, 256, 0, st>>>(A, lda, B, ldb, C, ldc, bias, M, N, K, act);
}

extern "C" void kernel_launch(void* const* d_in, const int* in_sizes, int n_in,
                              void* d_out, int out_size, void* d_ws, size_t ws_size,
                              hipStream_t stream)
{
    (void)in_sizes; (void)n_in; (void)out_size; (void)ws_size;
    const float* x_in = (const float*)d_in[0];
    MambaP m1 = { (const float*)d_in[1], (const float*)d_in[2], (const float*)d_in[3],
                  (const float*)d_in[4], (const float*)d_in[5], (const float*)d_in[6],
                  (const float*)d_in[7], (const float*)d_in[8], (const float*)d_in[9] };
    MambaP m2 = { (const float*)d_in[10], (const float*)d_in[11], (const float*)d_in[12],
                  (const float*)d_in[13], (const float*)d_in[14], (const float*)d_in[15],
                  (const float*)d_in[16], (const float*)d_in[17], (const float*)d_in[18] };
    const float* gate_w = (const float*)d_in[19];
    const float* moe_w1 = (const float*)d_in[20];
    const float* moe_b1 = (const float*)d_in[21];
    const float* moe_w2 = (const float*)d_in[22];
    const float* moe_b2 = (const float*)d_in[23];
    const float* q_w = (const float*)d_in[24];
    const float* k_w = (const float*)d_in[25];
    const float* v_w = (const float*)d_in[26];
    const float* o_w = (const float*)d_in[27];
    const float* f_w1 = (const float*)d_in[28];
    const float* f_b1 = (const float*)d_in[29];
    const float* ln_g = (const float*)d_in[30];
    const float* ln_b = (const float*)d_in[31];
    const float* f_w2 = (const float*)d_in[32];
    const float* f_b2 = (const float*)d_in[33];

    float* ws = (float*)d_ws;
    const size_t MF = 1024 * 1024;     // 1M floats
    float* X     = ws + 0 * MF;
    float* G     = ws + 1 * MF;
    float* G2    = ws + 2 * MF;
    float* G3    = ws + 3 * MF;
    float* T1    = ws + 4 * MF;
    float* ACC   = ws + 5 * MF;
    float* XZ    = ws + 6 * MF;        // 4M floats
    float* XI    = ws + 10 * MF;       // 2M
    float* DELTA = ws + 12 * MF;       // 2M
    float* Y     = ws + 14 * MF;       // 2M
    float* DBC   = ws + 16 * MF;       // 96K
    float* GATES = ws + 16 * MF + 131072;
    float* WTS   = ws + 16 * MF + 131072 + 8192;
    float* H1    = ws + 17 * MF;       // 4M
    float* Q     = ws + 21 * MF;
    float* KK    = ws + 22 * MF;
    float* VV    = ws + 22 * MF + 65536;
    float* AT    = ws + 23 * MF;
    float* XO    = ws + 24 * MF;

    const int NTOK = S_LEN * DIM_;     // 1M elems

    auto run_mamba = [&](const float* in, const MambaP& p, float* out) {
        gemm(stream, in, DIM_, p.in_w, 2 * D_INNER, XZ, 2 * D_INNER, nullptr,
             S_LEN, 2 * D_INNER, DIM_, 0);
        conv_silu_kernel<<<(S_LEN * D_INNER) / 256, 256, 0, stream>>>(XZ, p.conv_w, p.conv_b, XI);
        gemm(stream, XI, D_INNER, p.xproj_w, 96, DBC, 96, nullptr, S_LEN, 96, D_INNER, 0);
        gemm(stream, DBC, 96, p.dt_w, D_INNER, DELTA, D_INNER, p.dt_b,
             S_LEN, D_INNER, DT_RANK, 3);
        scan_kernel<<<D_INNER / 256, 256, 0, stream>>>(XI, DELTA, DBC, p.A_log, p.D, Y);
        gatemul_kernel<<<(S_LEN * D_INNER) / 256, 256, 0, stream>>>(Y, XZ);
        gemm(stream, Y, D_INNER, p.out_w, DIM_, out, DIM_, nullptr, S_LEN, DIM_, D_INNER, 0);
    };

    auto run_mml = [&](const float* in, float* out) {
        rms_kernel<<<S_LEN, 256, 0, stream>>>(in, G);
        run_mamba(G, m2, T1);
        add_kernel<<<NTOK / 256, 256, 0, stream>>>(G2, T1, G, NTOK);
        rms_kernel<<<S_LEN, 256, 0, stream>>>(G2, G3);
        gemm(stream, G3, DIM_, gate_w, NEXP, GATES, NEXP, nullptr, S_LEN, NEXP, DIM_, 0);
        gate_topk_kernel<<<S_LEN / 256, 256, 0, stream>>>(GATES, WTS);
        for (int e = 0; e < NEXP; ++e) {
            gemm(stream, G3, DIM_, moe_w1 + (size_t)e * DIM_ * HID_, HID_, H1, HID_,
                 moe_b1 + (size_t)e * HID_, S_LEN, HID_, DIM_, 2);
            gemm(stream, H1, HID_, moe_w2 + (size_t)e * HID_ * DIM_, DIM_, T1, DIM_,
                 moe_b2 + (size_t)e * DIM_, S_LEN, DIM_, HID_, 0);
            moe_accum_kernel<<<NTOK / 256, 256, 0, stream>>>(ACC, T1, WTS, e, e == 0);
        }
        add_kernel<<<NTOK / 256, 256, 0, stream>>>(out, ACC, in, NTOK);
    };

    auto run_tfm = [&](const float* in, float* out) {
        rms_kernel<<<S_LEN, 256, 0, stream>>>(in, G);
        gemm(stream, G, DIM_, q_w, DIM_, Q, DIM_, nullptr, S_LEN, DIM_, DIM_, 0);
        gemm(stream, G, DIM_, k_w, HDIM, KK, HDIM, nullptr, S_LEN, HDIM, DIM_, 0);
        gemm(stream, G, DIM_, v_w, HDIM, VV, HDIM, nullptr, S_LEN, HDIM, DIM_, 0);
        attn_kernel<<<dim3(S_LEN, NHEADS), 256, 0, stream>>>(Q, KK, VV, AT);
        gemm(stream, AT, DIM_, o_w, DIM_, T1, DIM_, nullptr, S_LEN, DIM_, DIM_, 0);
        add_kernel<<<NTOK / 256, 256, 0, stream>>>(G2, T1, in, NTOK);
        rms_kernel<<<S_LEN, 256, 0, stream>>>(G2, G3);
        gemm(stream, G3, DIM_, f_w1, HID_, H1, HID_, f_b1, S_LEN, HID_, DIM_, 1);
        ln_kernel<<<S_LEN, 256, 0, stream>>>(H1, ln_g, ln_b);
        gemm(stream, H1, HID_, f_w2, DIM_, T1, DIM_, f_b2, S_LEN, DIM_, HID_, 0);
        add_kernel<<<NTOK / 256, 256, 0, stream>>>(out, T1, G2, NTOK);
    };

    // x = mamba1(x); x = mml(x); x = tfm(x); x = mml(x); x = mamba1(x); out = mml(x)
    run_mamba(x_in, m1, X);
    run_mml(X, X);
    run_tfm(X, X);
    run_mml(X, X);
    run_mamba(X, m1, XO);
    run_mml(XO, (float*)d_out);
}
